// Encoder_overall_enhanced_33105607917951
// MI455X (gfx1250) — matrix operand order repaired and checked by reading, not compiled, not run on hardware
//
#include <hip/hip_runtime.h>

// ---------------------------------------------------------------------------
// Types for CDNA5 WMMA (wave32): v_wmma_f32_16x16x32_bf16
// ---------------------------------------------------------------------------
typedef __attribute__((ext_vector_type(16))) __bf16 bf16x16;
typedef __attribute__((ext_vector_type(8)))  float  floatx8;

struct U32x8 { uint4 a, b; };

#define TM 32
#define TN 64
#define TK 64
#define ASTR 80   // ushort stride per A-lds row (160B: 16B aligned, padded)
#define BSTR 80   // ushort stride per B-lds col (padded)

// ---- optional gfx1250 async global->LDS path (guarded; safe fallback) ----
#if defined(__has_builtin)
#if __has_builtin(__builtin_amdgcn_global_load_async_to_lds_b128)
#define HAVE_ASYNC_LDS 1
#endif
#endif
#ifndef HAVE_ASYNC_LDS
#define HAVE_ASYNC_LDS 0
#endif

#if HAVE_ASYNC_LDS
// Builtin signature (confirmed via compiler diagnostics):
//   void __builtin_amdgcn_global_load_async_to_lds_b128(
//        int4 addrspace(1)* src, int4 addrspace(3)* dst, imm int, imm int)
typedef int i32x4_v __attribute__((vector_size(16)));
#define AS1P(x) ((__attribute__((address_space(1))) i32x4_v*)(x))
#define AS3P(x) ((__attribute__((address_space(3))) i32x4_v*)(x))
static __device__ inline void wait_async0() {
#if __has_builtin(__builtin_amdgcn_s_wait_asynccnt)
  __builtin_amdgcn_s_wait_asynccnt(0);
#else
  asm volatile("s_wait_asynccnt 0" ::: "memory");
#endif
}
#endif

static __device__ inline unsigned short f2bfu(float f) {
  union { float f; unsigned u; } a; a.f = f;
  unsigned u = a.u;
  unsigned r = u + 0x7fffu + ((u >> 16) & 1u);   // round-to-nearest-even
  return (unsigned short)(r >> 16);
}

static __device__ inline uint2 pack4bf(float a, float b, float c, float d) {
  unsigned lo = (unsigned)f2bfu(a) | ((unsigned)f2bfu(b) << 16);
  unsigned hi = (unsigned)f2bfu(c) | ((unsigned)f2bfu(d) << 16);
  return make_uint2(lo, hi);
}

static __device__ inline float spikef(float x) {
  // spike(x) = floor(clamp(4x, 0, 4) + 0.5) * 0.25
  float t = 4.0f * x;
  t = fminf(fmaxf(t, 0.0f), 4.0f);
  return floorf(t + 0.5f) * 0.25f;
}

enum AMode { A_PLAIN = 0, A_FUSED2 = 1, A_SPIKE = 2, A_CONCAT2 = 3 };
enum Epi   { E_NONE = 0, E_SPIKE = 1, E_BIAS = 2, E_BIAS_SPIKE = 3, E_RESID = 4 };

// ---------------------------------------------------------------------------
// Generic bf16-WMMA GEMM: C[M,N] = epi(Aeff[M,K] @ B[K,N])
//   A_PLAIN  : Aeff = A0
//   A_FUSED2 : Aeff = cw[0]*A0 + cw[1]*A1 + cb[0]     (fused 1x1 conv)
//   A_SPIKE  : Aeff = spike(A0)
//   A_CONCAT2: Aeff = [A0 | A1], each row-stride 64, K = 128
// GK: K not a multiple of TK  (requires K % 4 == 0: per-float4 guard)
// GN: N not a multiple of TN  (requires N % 4 == 0: per-float4 guard)
// Block = 256 threads = 8 waves; each wave owns one 16x16 C tile of the
// 32x64 macro-tile; K-loop stages TKx(TM|TN) bf16 tiles through LDS.
// ---------------------------------------------------------------------------
template <int AMODE, int EPI, bool GK, bool GN>
__global__ __launch_bounds__(256)
void gemm_bf16_wmma(const float* __restrict__ A0, const float* __restrict__ A1,
                    const float* __restrict__ cw, const float* __restrict__ cb,
                    const float* __restrict__ B,  const float* __restrict__ bias,
                    const float* __restrict__ R,  float* __restrict__ C,
                    int M, int N, int K, int lda, int ldb, int ldc) {
  __shared__ alignas(16) unsigned short sA[TM * ASTR];
  __shared__ alignas(16) unsigned short sB[TN * BSTR];

  const int t    = threadIdx.x;
  const int lane = t & 31;
  const int wave = t >> 5;
  const int m0   = blockIdx.x * TM;
  const int n0   = blockIdx.y * TN;

  const int mi  = wave >> 2;   // 0..1 : 16-row sub-tile
  const int ni  = wave & 3;    // 0..3 : 16-col sub-tile
  const int hf  = lane >> 4;   // lane half (ISA layout)
  const int l16 = lane & 15;

  float w0 = 0.f, w1 = 0.f, b0 = 0.f;
  if (AMODE == A_FUSED2) { w0 = cw[0]; w1 = cw[1]; b0 = cb[0]; }

  floatx8 acc = {0.f, 0.f, 0.f, 0.f, 0.f, 0.f, 0.f, 0.f};

  for (int k0 = 0; k0 < K; k0 += TK) {
    // ------------ stage A tile (TM x TK) as bf16, row-major ----------------
    // vectorized: 2 x float4 per thread -> packed 4xbf16 b64 LDS stores.
    // GK: K % 4 == 0, so a float4 is fully in- or out-of-bounds.
    #pragma unroll
    for (int i = 0; i < (TM * TK) / (256 * 4); ++i) {
      int idx4 = (i * 256 + t) * 4;
      int row  = idx4 >> 6;          // / TK
      int kk   = idx4 & (TK - 1);
      float4 a4 = make_float4(0.f, 0.f, 0.f, 0.f);
      if (AMODE == A_FUSED2) {
        size_t off = (size_t)(m0 + row) * lda + k0 + kk;
        float4 p = *reinterpret_cast<const float4*>(&A0[off]);
        float4 q = *reinterpret_cast<const float4*>(&A1[off]);
        a4.x = w0 * p.x + w1 * q.x + b0;
        a4.y = w0 * p.y + w1 * q.y + b0;
        a4.z = w0 * p.z + w1 * q.z + b0;
        a4.w = w0 * p.w + w1 * q.w + b0;
      } else if (AMODE == A_CONCAT2) {
        const float* Ab = (k0 == 0) ? A0 : A1;   // TK==64 splits halves
        a4 = *reinterpret_cast<const float4*>(&Ab[(size_t)(m0 + row) * 64 + kk]);
      } else {
        if (!GK || (k0 + kk) < K) {
          a4 = *reinterpret_cast<const float4*>(&A0[(size_t)(m0 + row) * lda + k0 + kk]);
          if (AMODE == A_SPIKE) {
            a4.x = spikef(a4.x); a4.y = spikef(a4.y);
            a4.z = spikef(a4.z); a4.w = spikef(a4.w);
          }
        }
      }
      *reinterpret_cast<uint2*>(&sA[row * ASTR + kk]) =
          pack4bf(a4.x, a4.y, a4.z, a4.w);
    }
    // -------- stage B tile (TK x TN) as bf16, column-major [TN][TK] --------
    // vectorized: 4 x float4 per thread, bf16 scatter into transposed tile.
    // GK/GN: K % 4 == 0 and N % 4 == 0, so per-float4 guards suffice.
    #pragma unroll
    for (int i = 0; i < (TK * TN) / (256 * 4); ++i) {
      int idx4 = (i * 256 + t) * 4;
      int kk   = idx4 >> 6;          // / TN
      int nn   = idx4 & (TN - 1);
      int k    = k0 + kk;
      int n    = n0 + nn;
      bool ok = true;
      if (GK) ok = ok && (k < K);
      if (GN) ok = ok && (n < N);
      float4 b4 = make_float4(0.f, 0.f, 0.f, 0.f);
      if (ok) b4 = *reinterpret_cast<const float4*>(&B[(size_t)k * ldb + n]);
      sB[(nn + 0) * BSTR + kk] = f2bfu(b4.x);
      sB[(nn + 1) * BSTR + kk] = f2bfu(b4.y);
      sB[(nn + 2) * BSTR + kk] = f2bfu(b4.z);
      sB[(nn + 3) * BSTR + kk] = f2bfu(b4.w);
    }
    // speculative prefetch of next A stage (global_prefetch_b8)
    if (AMODE != A_CONCAT2) {
      if (k0 + TK < K) {
        int row = t >> 6, kk = t & 63;
        __builtin_prefetch(&A0[(size_t)(m0 + row) * lda + (k0 + TK) + kk], 0, 1);
        if (AMODE == A_FUSED2)
          __builtin_prefetch(&A1[(size_t)(m0 + row) * lda + (k0 + TK) + kk], 0, 1);
      }
    }
    __syncthreads();

    // ---------------- 2 x WMMA 16x16x32 over this TK=64 stage --------------
    #pragma unroll
    for (int kk2 = 0; kk2 < TK; kk2 += 32) {
      bf16x16 afrag, bfrag;
      {
        // A 16x32 bf16 layout: lane l (row M=l%16), elems 0..7 -> K=hf*8+e,
        // elems 8..15 -> K = 16 + hf*8 + (e-8)
        int row = mi * 16 + l16;
        int kb  = kk2 + hf * 8;
        const unsigned short* pa = &sA[row * ASTR + kb];
        U32x8 tmp;
        tmp.a = *reinterpret_cast<const uint4*>(pa);        // K = kb .. kb+7
        tmp.b = *reinterpret_cast<const uint4*>(pa + 16);   // K = kb+16 .. kb+23
        __builtin_memcpy(&afrag, &tmp, 32);
      }
      {
        // B 32x16 bf16 layout: lane l (col N=l%16), elems 0..7 -> K=hf*8+e,
        // elems 8..15 -> K = 16 + hf*8 + (e-8)
        int col = ni * 16 + l16;
        int kb  = kk2 + hf * 8;
        const unsigned short* pb = &sB[col * BSTR + kb];
        U32x8 tmp;
        tmp.a = *reinterpret_cast<const uint4*>(pb);
        tmp.b = *reinterpret_cast<const uint4*>(pb + 16);
        __builtin_memcpy(&bfrag, &tmp, 32);
      }
      acc = __builtin_amdgcn_wmma_f32_16x16x32_bf16(
          /*neg_a=*/false, afrag, /*neg_b=*/false, bfrag,
          /*c_mod=*/(short)0, acc, /*reuse_a=*/false, /*reuse_b=*/false);
    }
    __syncthreads();
  }

  // ---- epilogue: C layout -> VGPR r: row = r + 8*hf, col = l16 ----
  int col = n0 + ni * 16 + l16;
  if (!GN || col < N) {
    float bv = 0.0f;
    if (EPI == E_BIAS || EPI == E_BIAS_SPIKE) bv = bias[col];
    #pragma unroll
    for (int r = 0; r < 8; ++r) {
      int row = m0 + mi * 16 + r + hf * 8;
      float v = acc[r] + bv;
      if (EPI == E_SPIKE || EPI == E_BIAS_SPIKE) v = spikef(v);
      if (EPI == E_RESID) v += R[(size_t)row * ldc + col];
      C[(size_t)row * ldc + col] = v;
    }
  }
}

// ---------------------------------------------------------------------------
// kv[h,d,e] = sum_n k[n, h*8+d] * v[n, h*8+e]   (H=8, d=e=8 -> 512 values)
// Partial per 128-row block (48 blocks), deterministic reduce after.
// k/v tiles staged via async global->LDS when the toolchain supports it.
// ---------------------------------------------------------------------------
__global__ __launch_bounds__(256)
void kv_partial(const float* __restrict__ Kq, const float* __restrict__ Vq,
                float* __restrict__ part) {
  __shared__ alignas(16) float sK[128][64];
  __shared__ alignas(16) float sV[128][64];
  int t  = threadIdx.x;
  int r0 = blockIdx.x * 128;
#if HAVE_ASYNC_LDS
  #pragma unroll
  for (int i = 0; i < 8; ++i) {              // 128*64 floats / (256 thr * 4)
    int idx4 = (i * 256 + t) * 4;
    int r = idx4 >> 6, c = idx4 & 63;
    __builtin_amdgcn_global_load_async_to_lds_b128(
        AS1P(Kq + (size_t)(r0 + r) * 64 + c), AS3P(&sK[r][c]), 0, 0);
    __builtin_amdgcn_global_load_async_to_lds_b128(
        AS1P(Vq + (size_t)(r0 + r) * 64 + c), AS3P(&sV[r][c]), 0, 0);
  }
  wait_async0();
#else
  #pragma unroll
  for (int i = 0; i < 8; ++i) {
    int idx4 = (i * 256 + t) * 4;
    int r = idx4 >> 6, c = idx4 & 63;
    *reinterpret_cast<float4*>(&sK[r][c]) =
        *reinterpret_cast<const float4*>(&Kq[(size_t)(r0 + r) * 64 + c]);
    *reinterpret_cast<float4*>(&sV[r][c]) =
        *reinterpret_cast<const float4*>(&Vq[(size_t)(r0 + r) * 64 + c]);
  }
#endif
  __syncthreads();
  #pragma unroll
  for (int j = 0; j < 2; ++j) {
    int idx = j * 256 + t;                   // h*64 + d*8 + e
    int h = idx >> 6, d = (idx >> 3) & 7, e = idx & 7;
    float acc = 0.f;
    for (int r = 0; r < 128; ++r)
      acc += sK[r][h * 8 + d] * sV[r][h * 8 + e];
    part[(size_t)blockIdx.x * 512 + idx] = acc;
  }
}

__global__ void kv_reduce(const float* __restrict__ part, float* __restrict__ kv,
                          int nparts) {
  int idx = blockIdx.x * blockDim.x + threadIdx.x;   // 0..511
  float acc = 0.f;
  for (int p = 0; p < nparts; ++p) acc += part[(size_t)p * 512 + idx];
  kv[idx] = acc;
}

// o[n, h*8+e] = 0.25 * sum_d q[n, h*8+d] * kv[h,d,e]
__global__ __launch_bounds__(256)
void attn_o(const float* __restrict__ q, const float* __restrict__ kv,
            float* __restrict__ o) {
  __shared__ alignas(16) float skv[512];
  int t = threadIdx.x;
#if HAVE_ASYNC_LDS
  if (t < 128)
    __builtin_amdgcn_global_load_async_to_lds_b128(
        AS1P(kv + t * 4), AS3P(&skv[t * 4]), 0, 0);
  wait_async0();
#else
  skv[t] = kv[t];
  skv[t + 256] = kv[t + 256];
#endif
  __syncthreads();
  int gid = blockIdx.x * 256 + t;
  int n = gid >> 6, c = gid & 63;
  int h = c >> 3, e = c & 7;
  float acc = 0.f;
  #pragma unroll
  for (int d = 0; d < 8; ++d)
    acc += q[(size_t)n * 64 + h * 8 + d] * skv[h * 64 + d * 8 + e];
  o[gid] = 0.25f * acc;
}

__global__ void spike_copy(const float* __restrict__ in, float* __restrict__ out) {
  int gid = blockIdx.x * blockDim.x + threadIdx.x;
  out[gid] = spikef(in[gid]);
}

// ---------------------------------------------------------------------------
// Host launcher
// ---------------------------------------------------------------------------
extern "C" void kernel_launch(void* const* d_in, const int* in_sizes, int n_in,
                              void* d_out, int out_size, void* d_ws, size_t ws_size,
                              hipStream_t stream) {
  const int Nn = 6144, DIN1 = 3000, DIN2 = 512, DOUT = 64;

  const float* feat1   = (const float*)d_in[0];
  const float* feat2   = (const float*)d_in[1];
  const float* adj_sp1 = (const float*)d_in[2];
  const float* adj_ft1 = (const float*)d_in[3];
  const float* adj_sp2 = (const float*)d_in[4];
  const float* adj_ft2 = (const float*)d_in[5];
  const float* c1w = (const float*)d_in[6];
  const float* c1b = (const float*)d_in[7];
  const float* c2w = (const float*)d_in[8];
  const float* c2b = (const float*)d_in[9];
  const float* w_enc1 = (const float*)d_in[10];
  const float* wq1 = (const float*)d_in[11];
  const float* wk1 = (const float*)d_in[12];
  const float* wv1 = (const float*)d_in[13];
  const float* wp1 = (const float*)d_in[14];
  const float* w_enc2 = (const float*)d_in[15];
  const float* wq2 = (const float*)d_in[16];
  const float* wk2 = (const float*)d_in[17];
  const float* wv2 = (const float*)d_in[18];
  const float* wp2 = (const float*)d_in[19];
  const float* fc1_w = (const float*)d_in[20];
  const float* fc1_b = (const float*)d_in[21];
  const float* fc2_w = (const float*)d_in[22];
  const float* fc2_b = (const float*)d_in[23];
  const float* w_dec1 = (const float*)d_in[24];
  const float* w_dec2 = (const float*)d_in[25];

  const size_t NV = (size_t)Nn * DOUT;     // 393216 floats
  float* ws = (float*)d_ws;
  float* fe1  = ws + 0 * NV;
  float* fe2  = ws + 1 * NV;
  float* x1   = ws + 2 * NV;
  float* x2   = ws + 3 * NV;
  float* qb   = ws + 4 * NV;
  float* kb   = ws + 5 * NV;
  float* vb   = ws + 6 * NV;
  float* ob   = ws + 7 * NV;
  float* hb   = ws + 8 * NV;
  float* comb = ws + 9 * NV;
  float* gb   = ws + 10 * NV;
  float* kvp  = ws + 11 * NV;              // 48*512
  float* kvb  = kvp + 48 * 512;            // 512

  float* out    = (float*)d_out;
  float* emb1   = out;
  float* emb2   = out + NV;
  float* scomb  = out + 2 * NV;
  float* recon1 = out + 3 * NV;
  float* recon2 = recon1 + (size_t)Nn * DIN1;

  dim3 blk(256);
  dim3 gM(Nn / TM, 1);                     // 192 x 1 (N = 64)
  dim3 gD1(Nn / TM, (DIN1 + TN - 1) / TN); // 192 x 47
  dim3 gD2(Nn / TM, DIN2 / TN);            // 192 x 8

  // --- feature embeddings: fe = feat @ w_enc ---
  gemm_bf16_wmma<A_PLAIN, E_NONE, true, false><<<gM, blk, 0, stream>>>(
      feat1, nullptr, nullptr, nullptr, w_enc1, nullptr, nullptr, fe1,
      Nn, DOUT, DIN1, DIN1, DOUT, DOUT);
  gemm_bf16_wmma<A_PLAIN, E_NONE, false, false><<<gM, blk, 0, stream>>>(
      feat2, nullptr, nullptr, nullptr, w_enc2, nullptr, nullptr, fe2,
      Nn, DOUT, DIN2, DIN2, DOUT, DOUT);

  // ===== encoder 1 =====
  gemm_bf16_wmma<A_FUSED2, E_SPIKE, false, false><<<gM, blk, 0, stream>>>(
      adj_sp1, adj_ft1, c1w, c1b, fe1, nullptr, nullptr, x1,
      Nn, DOUT, Nn, Nn, DOUT, DOUT);
  gemm_bf16_wmma<A_SPIKE, E_SPIKE, false, false><<<gM, blk, 0, stream>>>(
      x1, nullptr, nullptr, nullptr, wq1, nullptr, nullptr, qb,
      Nn, DOUT, DOUT, DOUT, DOUT, DOUT);
  gemm_bf16_wmma<A_SPIKE, E_SPIKE, false, false><<<gM, blk, 0, stream>>>(
      x1, nullptr, nullptr, nullptr, wk1, nullptr, nullptr, kb,
      Nn, DOUT, DOUT, DOUT, DOUT, DOUT);
  gemm_bf16_wmma<A_SPIKE, E_SPIKE, false, false><<<gM, blk, 0, stream>>>(
      x1, nullptr, nullptr, nullptr, wv1, nullptr, nullptr, vb,
      Nn, DOUT, DOUT, DOUT, DOUT, DOUT);
  kv_partial<<<dim3(48), blk, 0, stream>>>(kb, vb, kvp);
  kv_reduce<<<dim3(2), blk, 0, stream>>>(kvp, kvb, 48);
  attn_o<<<dim3((Nn * DOUT) / 256), blk, 0, stream>>>(qb, kvb, ob);
  gemm_bf16_wmma<A_PLAIN, E_RESID, false, false><<<gM, blk, 0, stream>>>(
      ob, nullptr, nullptr, nullptr, wp1, nullptr, x1, emb1,
      Nn, DOUT, DOUT, DOUT, DOUT, DOUT);

  // ===== encoder 2 =====
  gemm_bf16_wmma<A_FUSED2, E_SPIKE, false, false><<<gM, blk, 0, stream>>>(
      adj_sp2, adj_ft2, c2w, c2b, fe2, nullptr, nullptr, x2,
      Nn, DOUT, Nn, Nn, DOUT, DOUT);
  gemm_bf16_wmma<A_SPIKE, E_SPIKE, false, false><<<gM, blk, 0, stream>>>(
      x2, nullptr, nullptr, nullptr, wq2, nullptr, nullptr, qb,
      Nn, DOUT, DOUT, DOUT, DOUT, DOUT);
  gemm_bf16_wmma<A_SPIKE, E_SPIKE, false, false><<<gM, blk, 0, stream>>>(
      x2, nullptr, nullptr, nullptr, wk2, nullptr, nullptr, kb,
      Nn, DOUT, DOUT, DOUT, DOUT, DOUT);
  gemm_bf16_wmma<A_SPIKE, E_SPIKE, false, false><<<gM, blk, 0, stream>>>(
      x2, nullptr, nullptr, nullptr, wv2, nullptr, nullptr, vb,
      Nn, DOUT, DOUT, DOUT, DOUT, DOUT);
  kv_partial<<<dim3(48), blk, 0, stream>>>(kb, vb, kvp);
  kv_reduce<<<dim3(2), blk, 0, stream>>>(kvp, kvb, 48);
  attn_o<<<dim3((Nn * DOUT) / 256), blk, 0, stream>>>(qb, kvb, ob);
  gemm_bf16_wmma<A_PLAIN, E_RESID, false, false><<<gM, blk, 0, stream>>>(
      ob, nullptr, nullptr, nullptr, wp2, nullptr, x2, emb2,
      Nn, DOUT, DOUT, DOUT, DOUT, DOUT);

  // ===== fused MLP: comb = spike([emb1|emb2] @ fc1 + b1) @ fc2 + b2 =====
  gemm_bf16_wmma<A_CONCAT2, E_BIAS_SPIKE, false, false><<<gM, blk, 0, stream>>>(
      emb1, emb2, nullptr, nullptr, fc1_w, fc1_b, nullptr, hb,
      Nn, DOUT, 2 * DOUT, DOUT, DOUT, DOUT);
  gemm_bf16_wmma<A_PLAIN, E_BIAS, false, false><<<gM, blk, 0, stream>>>(
      hb, nullptr, nullptr, nullptr, fc2_w, fc2_b, nullptr, comb,
      Nn, DOUT, DOUT, DOUT, DOUT, DOUT);
  spike_copy<<<dim3((Nn * DOUT) / 256), blk, 0, stream>>>(comb, scomb);

  // ===== decoders (re-associated): recon = spike((adj_sp @ comb) @ w_dec) ====
  gemm_bf16_wmma<A_PLAIN, E_NONE, false, false><<<gM, blk, 0, stream>>>(
      adj_sp1, nullptr, nullptr, nullptr, comb, nullptr, nullptr, gb,
      Nn, DOUT, Nn, Nn, DOUT, DOUT);
  gemm_bf16_wmma<A_PLAIN, E_SPIKE, false, true><<<gD1, blk, 0, stream>>>(
      gb, nullptr, nullptr, nullptr, w_dec1, nullptr, nullptr, recon1,
      Nn, DIN1, DOUT, DOUT, DIN1, DIN1);

  gemm_bf16_wmma<A_PLAIN, E_NONE, false, false><<<gM, blk, 0, stream>>>(
      adj_sp2, nullptr, nullptr, nullptr, comb, nullptr, nullptr, gb,
      Nn, DOUT, Nn, Nn, DOUT, DOUT);
  gemm_bf16_wmma<A_PLAIN, E_SPIKE, false, false><<<gD2, blk, 0, stream>>>(
      gb, nullptr, nullptr, nullptr, w_dec2, nullptr, nullptr, recon2,
      Nn, DIN2, DOUT, DOUT, DIN2, DIN2);
}
